// TNODE_25417616458373
// MI455X (gfx1250) — matrix operand order repaired and checked by reading, not compiled, not run on hardware
//
#include <hip/hip_runtime.h>

typedef __attribute__((ext_vector_type(16))) _Float16 v16h;
typedef __attribute__((ext_vector_type(8)))  float    v8f;

#define N_CELLS 8192
#define G_GENES 2000
#define L_LAT   20
#define H_ODE   128
#define H_VAE   128
#define KCHUNKS 63            // ceil(2000/32); last chunk is half-valid (K=1984..1999)

// workspace layout (float elements); ws base is >=256B aligned from the harness
#define WS_TS    0
#define WS_ZS    (WS_TS + N_CELLS)
#define WS_ZU    (WS_ZS + N_CELLS * L_LAT)
#define WS_TSS   (WS_ZU + N_CELLS * L_LAT)
#define WS_IDX   (WS_TSS + N_CELLS)
#define WS_PREDZ (WS_IDX + N_CELLS)
#define WS_ACC   (WS_PREDZ + N_CELLS * L_LAT)       // acc[0]=ec,[1]=ecu,[2]=zdiv,[3]=xt
#define WS_BE    ((WS_ACC + 4 + 7) & ~7)            // 32B-aligned; f16 packed W_e1 B-fragments
#define BE_HALF  (KCHUNKS * H_VAE * 32)             // 63*128*32 halves
#define WS_BD    (WS_BE + BE_HALF / 2)              // f16 packed W_d2 B-fragments
#define BD_HALF  (G_GENES * 4 * 32)                 // 2000*4*32 halves

__device__ __forceinline__ float eluf(float x) { return x > 0.f ? x : __expf(x) - 1.f; }

__device__ __forceinline__ void cvt8(v16h& a, int base, float4 lo, float4 hi) {
    a[base + 0] = (_Float16)lo.x; a[base + 1] = (_Float16)lo.y;
    a[base + 2] = (_Float16)lo.z; a[base + 3] = (_Float16)lo.w;
    a[base + 4] = (_Float16)hi.x; a[base + 5] = (_Float16)hi.y;
    a[base + 6] = (_Float16)hi.z; a[base + 7] = (_Float16)hi.w;
}

// ---------------------------------------------------------------------------
// zero the loss accumulators (harness poisons ws; must re-zero every call)
// ---------------------------------------------------------------------------
__global__ void zero_acc_kernel(float* acc) {
    if (threadIdx.x < 4) acc[threadIdx.x] = 0.f;
}

// ---------------------------------------------------------------------------
// One-time weight packing: f32 -> f16 in WMMA B-fragment order so the GEMM
// loops load one contiguous 32B v16h per lane (2x global_load_b128) instead
// of 16 strided scalar gathers.
// B layout (16-bit 32x16): lane col = lane&15, half = lane>>4 selects K block.
// ---------------------------------------------------------------------------
__global__ void pack_We1_kernel(const float* __restrict__ We1, _Float16* __restrict__ out) {
    int id = blockIdx.x * 256 + threadIdx.x;          // KCHUNKS*128*2 jobs
    if (id >= KCHUNKS * H_VAE * 2) return;
    int h = id & 1;
    int n = (id >> 1) & (H_VAE - 1);
    int c = id >> 8;
    _Float16* dst = out + ((size_t)(c * H_VAE + n)) * 32 + h * 16;
    int kbase = c * 32 + h * 8;
#pragma unroll
    for (int e = 0; e < 16; ++e) {
        int k = kbase + (e < 8 ? e : 8 + e);
        dst[e] = (_Float16)(k < G_GENES ? We1[(size_t)k * H_VAE + n] : 0.f);
    }
}

__global__ void pack_Wd2_kernel(const float* __restrict__ Wd2, _Float16* __restrict__ out) {
    int id = blockIdx.x * 256 + threadIdx.x;          // 2000*4*2 jobs
    if (id >= G_GENES * 4 * 2) return;
    int h  = id & 1;
    int kc = (id >> 1) & 3;
    int n  = id >> 3;
    _Float16* dst = out + ((size_t)(n * 4 + kc)) * 32 + h * 16;
    int kbase = kc * 32 + h * 8;
#pragma unroll
    for (int e = 0; e < 16; ++e)
        dst[e] = (_Float16)Wd2[(size_t)(kbase + (e < 8 ? e : 8 + e)) * G_GENES + n];
}

// ---------------------------------------------------------------------------
// Encoder: h = elu(X @ W_e1 + b_e1) [16 rows/WG, 8 waves each own a 16-col
// tile; K-loop branch-free with float4 A loads + packed f16 B fragments],
// then z = h @ W_ez + b_ez and Ts = sigmoid(h @ W_et + b_et) via VALU on LDS.
// ---------------------------------------------------------------------------
__global__ __launch_bounds__(256)
void encoder_kernel(const float* __restrict__ X, const _Float16* __restrict__ BE,
                    const float* __restrict__ be1,
                    const float* __restrict__ Wet, const float* __restrict__ bet,
                    const float* __restrict__ Wez, const float* __restrict__ bez,
                    float* __restrict__ zout, float* __restrict__ Tsout, int writeTs) {
    __shared__ float hbuf[16 * H_VAE];
    const int tid  = threadIdx.x;
    const int wv   = tid >> 5;
    const int lane = tid & 31;
    const int m16  = lane & 15;
    const int half = lane >> 4;
    const int r0   = blockIdx.x * 16;
    const int ncol = wv * 16 + m16;

    const float*    arow = X + (size_t)(r0 + m16) * G_GENES;
    const _Float16* bfr  = BE + (size_t)ncol * 32 + half * 16;

    v8f c = {};
    // 62 full K-chunks: no bounds checks, vector loads only
    for (int ch = 0; ch < KCHUNKS - 1; ++ch) {
        const int kc = ch * 32;
        float4 x0 = *(const float4*)(arow + kc + half * 8);
        float4 x1 = *(const float4*)(arow + kc + half * 8 + 4);
        float4 y0 = *(const float4*)(arow + kc + 16 + half * 8);
        float4 y1 = *(const float4*)(arow + kc + 16 + half * 8 + 4);
        v16h a;
        cvt8(a, 0, x0, x1);
        cvt8(a, 8, y0, y1);
        v16h b = *(const v16h*)(bfr + (size_t)ch * H_VAE * 32);
        c = __builtin_amdgcn_wmma_f32_16x16x32_f16(false, a, false, b, (short)0, c, false, false);
    }
    {   // tail chunk kc=1984: only K=1984..1999 valid -> upper half of A is zero
        const int kc = (KCHUNKS - 1) * 32;
        float4 x0 = *(const float4*)(arow + kc + half * 8);
        float4 x1 = *(const float4*)(arow + kc + half * 8 + 4);
        v16h a;
        cvt8(a, 0, x0, x1);
#pragma unroll
        for (int e = 8; e < 16; ++e) a[e] = (_Float16)0.f;
        v16h b = *(const v16h*)(bfr + (size_t)(KCHUNKS - 1) * H_VAE * 32);  // pre-zero-padded
        c = __builtin_amdgcn_wmma_f32_16x16x32_f16(false, a, false, b, (short)0, c, false, false);
    }
    // C layout: VGPR r, lane -> row m = r + 8*half, col = lane&15
    const float bias = be1[ncol];
#pragma unroll
    for (int r = 0; r < 8; ++r) {
        int m = r + 8 * half;
        hbuf[m * H_VAE + ncol] = eluf(c[r] + bias);
    }
    __syncthreads();

    // small heads: 16 rows x (20 latent + 1 time) dot products of length 128
    for (int job = tid; job < 16 * (L_LAT + 1); job += 256) {
        int row = job / (L_LAT + 1);
        int o   = job % (L_LAT + 1);
        const float* h = hbuf + row * H_VAE;
        if (o < L_LAT) {
            float acc = bez[o];
            for (int k = 0; k < H_VAE; ++k) acc += h[k] * Wez[k * L_LAT + o];
            zout[(size_t)(r0 + row) * L_LAT + o] = acc;
        } else if (writeTs) {
            float acc = bet[0];
            for (int k = 0; k < H_VAE; ++k) acc += h[k] * Wet[k];
            Tsout[r0 + row] = 1.f / (1.f + __expf(-acc));
        }
    }
}

// ---------------------------------------------------------------------------
// Single-workgroup bitonic argsort of 8192 positive floats.
// Key packed as (float_bits << 32) | index -> deterministic tie-break.
// ---------------------------------------------------------------------------
__global__ __launch_bounds__(1024)
void sort_kernel(const float* __restrict__ Ts, float* __restrict__ Tss, int* __restrict__ idxs) {
    __shared__ unsigned long long kv[N_CELLS];
    const int tid = threadIdx.x;
    for (int i = tid; i < N_CELLS; i += 1024)
        kv[i] = ((unsigned long long)__float_as_uint(Ts[i]) << 32) | (unsigned int)i;
    for (int k = 2; k <= N_CELLS; k <<= 1) {
        for (int j = k >> 1; j > 0; j >>= 1) {
            __syncthreads();
            for (int i = tid; i < N_CELLS; i += 1024) {
                int ixj = i ^ j;
                if (ixj > i) {
                    unsigned long long va = kv[i], vb = kv[ixj];
                    bool up = ((i & k) == 0);
                    if ((va > vb) == up) { kv[i] = vb; kv[ixj] = va; }
                }
            }
        }
    }
    __syncthreads();
    for (int i = tid; i < N_CELLS; i += 1024) {
        unsigned long long v = kv[i];
        idxs[i] = (int)(v & 0xffffffffu);
        Tss[i]  = __uint_as_float((unsigned int)(v >> 32));
    }
}

// ---------------------------------------------------------------------------
// Sequential Euler scan over the sorted time grid: one wave32, 8191 steps.
// Hidden dim 128 split 4 units/lane; W_o1 + b in registers, W_o2 in LDS;
// cross-lane butterfly reduce with shfl_xor (no barriers: single wave).
// ---------------------------------------------------------------------------
__global__ __launch_bounds__(32)
void ode_kernel(const float* __restrict__ Tss, const int* __restrict__ idxs,
                const float* __restrict__ zs,
                const float* __restrict__ Wo1, const float* __restrict__ bo1,
                const float* __restrict__ Wo2, const float* __restrict__ bo2,
                float* __restrict__ predz) {
    __shared__ float W2[H_ODE * L_LAT];
    const int lane = threadIdx.x;
    for (int i = lane; i < H_ODE * L_LAT; i += 32) W2[i] = Wo2[i];
    __syncthreads();

    float w1[4][L_LAT], b1[4];
#pragma unroll
    for (int q = 0; q < 4; ++q) {
        int i = lane + q * 32;
        b1[q] = bo1[i];
#pragma unroll
        for (int j = 0; j < L_LAT; ++j) w1[q][j] = Wo1[j * H_ODE + i];
    }
    float b2l[L_LAT], z[L_LAT];
    const int i0 = idxs[0];
#pragma unroll
    for (int j = 0; j < L_LAT; ++j) { b2l[j] = bo2[j]; z[j] = zs[(size_t)i0 * L_LAT + j]; }

    for (int t = 0; t < N_CELLS; ++t) {
        float myz = 0.f;
#pragma unroll
        for (int j = 0; j < L_LAT; ++j) myz = (lane == j) ? z[j] : myz;
        if (lane < L_LAT) predz[(size_t)t * L_LAT + lane] = myz;
        if (t == N_CELLS - 1) break;

        const float dt = Tss[t + 1] - Tss[t];
        float p[L_LAT];
#pragma unroll
        for (int j = 0; j < L_LAT; ++j) p[j] = 0.f;
#pragma unroll
        for (int q = 0; q < 4; ++q) {
            float h = b1[q];
#pragma unroll
            for (int j = 0; j < L_LAT; ++j) h += z[j] * w1[q][j];
            float e = eluf(h);
            const int i = lane + q * 32;
#pragma unroll
            for (int j = 0; j < L_LAT; ++j) p[j] += e * W2[i * L_LAT + j];
        }
#pragma unroll
        for (int m = 1; m < 32; m <<= 1) {
#pragma unroll
            for (int j = 0; j < L_LAT; ++j) p[j] += __shfl_xor(p[j], m, 32);
        }
#pragma unroll
        for (int j = 0; j < L_LAT; ++j) z[j] += dt * (p[j] + b2l[j]);
    }
}

// ---------------------------------------------------------------------------
// z_div + recon_xt: one wave per sorted row; f(pred_z) via ODE-MLP (weights in
// LDS), target gathered through idx; means accumulated atomically.
// ---------------------------------------------------------------------------
__global__ __launch_bounds__(256)
void zdiv_kernel(const float* __restrict__ predz, const int* __restrict__ idxs,
                 const float* __restrict__ zs, const float* __restrict__ zu,
                 const float* __restrict__ Wo1, const float* __restrict__ bo1,
                 const float* __restrict__ Wo2, const float* __restrict__ bo2,
                 const float* __restrict__ beta, const float* __restrict__ lam,
                 float* __restrict__ acc) {
    __shared__ float W1[L_LAT * H_ODE], W2[H_ODE * L_LAT], B1[H_ODE];
    const int tid = threadIdx.x;
    for (int i = tid; i < L_LAT * H_ODE; i += 256) { W1[i] = Wo1[i]; W2[i] = Wo2[i]; }
    for (int i = tid; i < H_ODE; i += 256) B1[i] = bo1[i];
    __syncthreads();

    const int wv = tid >> 5, lane = tid & 31;
    const int n  = blockIdx.x * 8 + wv;
    float z[L_LAT];
#pragma unroll
    for (int j = 0; j < L_LAT; ++j) z[j] = predz[(size_t)n * L_LAT + j];

    float p[L_LAT];
#pragma unroll
    for (int j = 0; j < L_LAT; ++j) p[j] = 0.f;
#pragma unroll
    for (int q = 0; q < 4; ++q) {
        const int i = lane + q * 32;
        float h = B1[i];
#pragma unroll
        for (int j = 0; j < L_LAT; ++j) h += z[j] * W1[j * H_ODE + i];
        float e = eluf(h);
#pragma unroll
        for (int j = 0; j < L_LAT; ++j) p[j] += e * W2[i * L_LAT + j];
    }
#pragma unroll
    for (int m = 1; m < 32; m <<= 1) {
#pragma unroll
        for (int j = 0; j < L_LAT; ++j) p[j] += __shfl_xor(p[j], m, 32);
    }

    const int src = idxs[n];
    float szdiv = 0.f, sxt = 0.f;
#pragma unroll
    for (int j = 0; j < L_LAT; ++j) {
        float zsj = zs[(size_t)src * L_LAT + j];
        float zuj = zu[(size_t)src * L_LAT + j];
        float target = __expf(beta[j]) * zuj - __expf(lam[j]) * zsj;
        float fj = p[j] + bo2[j];
        float d1 = fj - target; szdiv += d1 * d1;
        float d2 = zsj - z[j];  sxt   += d2 * d2;
    }
    if (lane == 0) { atomicAdd(&acc[2], szdiv); atomicAdd(&acc[3], sxt); }
}

// ---------------------------------------------------------------------------
// Decoder + fused squared-error reduction (permutation-invariant, unsorted):
// hd = elu(z @ W_d1 + b_d1) on 16 rows (VALU), then WMMA over the 2000-wide
// output in 16-col tiles (K=128 = 4 x 16x16x32) with packed f16 B fragments;
// pred_x never materialized.
// ---------------------------------------------------------------------------
__global__ __launch_bounds__(256)
void decoder_kernel(const float* __restrict__ Z, const float* __restrict__ X,
                    const float* __restrict__ Wd1, const float* __restrict__ bd1,
                    const _Float16* __restrict__ BD, const float* __restrict__ bd2,
                    float* __restrict__ acc, int accIdx) {
    __shared__ float zl[16 * L_LAT];
    __shared__ float hd[16 * H_VAE];
    __shared__ float red[256];
    const int tid = threadIdx.x;
    const int r0  = blockIdx.x * 16;

    for (int i = tid; i < 16 * L_LAT; i += 256) zl[i] = Z[(size_t)r0 * L_LAT + i];
    __syncthreads();
    for (int job = tid; job < 16 * H_VAE; job += 256) {
        int row = job >> 7, i = job & (H_VAE - 1);
        float a = bd1[i];
#pragma unroll
        for (int j = 0; j < L_LAT; ++j) a += zl[row * L_LAT + j] * Wd1[j * H_VAE + i];
        hd[job] = eluf(a);
    }
    __syncthreads();

    const int wv = tid >> 5, lane = tid & 31;
    const int m16 = lane & 15, half = lane >> 4;
    // A fragments fixed per wave: hd is 16 x 128 (K), 4 chunks of K=32
    v16h afr[4];
#pragma unroll
    for (int kc = 0; kc < 4; ++kc) {
#pragma unroll
        for (int e = 0; e < 8; ++e) {
            afr[kc][e]     = (_Float16)hd[m16 * H_VAE + kc * 32 + half * 8 + e];
            afr[kc][e + 8] = (_Float16)hd[m16 * H_VAE + kc * 32 + 16 + half * 8 + e];
        }
    }
    float myacc = 0.f;
    for (int ct = wv; ct < G_GENES / 16; ct += 8) {
        const int ncol = ct * 16 + m16;
        v8f c = {};
#pragma unroll
        for (int kc = 0; kc < 4; ++kc) {
            v16h b = *(const v16h*)(BD + ((size_t)(ncol * 4 + kc)) * 32 + half * 16);
            c = __builtin_amdgcn_wmma_f32_16x16x32_f16(false, afr[kc], false, b, (short)0, c, false, false);
        }
        const float bias = bd2[ncol];
#pragma unroll
        for (int r = 0; r < 8; ++r) {
            int m = r + 8 * half;
            float d = X[(size_t)(r0 + m) * G_GENES + ncol] - (c[r] + bias);
            myacc += d * d;
        }
    }
    red[tid] = myacc;
    __syncthreads();
    if (tid == 0) {
        float s = 0.f;
        for (int i = 0; i < 256; ++i) s += red[i];
        atomicAdd(&acc[accIdx], s);
    }
}

// ---------------------------------------------------------------------------
// Finalize: means + total loss. Output order: (loss, ec, ec_u, z_div).
// ---------------------------------------------------------------------------
__global__ void finalize_kernel(const float* __restrict__ acc, float* __restrict__ out) {
    if (threadIdx.x == 0) {
        const float ng = (float)((long long)N_CELLS * G_GENES);
        const float nl = (float)(N_CELLS * L_LAT);
        float ec   = acc[0] / ng;
        float ecu  = acc[1] / ng;
        float zdiv = acc[2] / nl;
        float xt   = acc[3] / nl;
        out[0] = ec + ecu + zdiv + xt;
        out[1] = ec;
        out[2] = ecu;
        out[3] = zdiv;
    }
}

extern "C" void kernel_launch(void* const* d_in, const int* in_sizes, int n_in,
                              void* d_out, int out_size, void* d_ws, size_t ws_size,
                              hipStream_t stream) {
    (void)in_sizes; (void)n_in; (void)out_size; (void)ws_size;
    const float* s   = (const float*)d_in[0];
    const float* u   = (const float*)d_in[1];
    // d_in[2] = Pt, d_in[3] = g : unused by the reference loss (pt_loss = 0)
    const float* We1 = (const float*)d_in[4];
    const float* be1 = (const float*)d_in[5];
    const float* Wet = (const float*)d_in[6];
    const float* bet = (const float*)d_in[7];
    const float* Wez = (const float*)d_in[8];
    const float* bez = (const float*)d_in[9];
    const float* Wo1 = (const float*)d_in[10];
    const float* bo1 = (const float*)d_in[11];
    const float* Wo2 = (const float*)d_in[12];
    const float* bo2 = (const float*)d_in[13];
    const float* Wd1 = (const float*)d_in[14];
    const float* bd1 = (const float*)d_in[15];
    const float* Wd2 = (const float*)d_in[16];
    const float* bd2 = (const float*)d_in[17];
    const float* beta = (const float*)d_in[18];
    const float* lam  = (const float*)d_in[19];

    float* ws    = (float*)d_ws;
    float* Ts    = ws + WS_TS;
    float* zsb   = ws + WS_ZS;
    float* zub   = ws + WS_ZU;
    float* Tss   = ws + WS_TSS;
    int*   idxs  = (int*)(ws + WS_IDX);
    float* predz = ws + WS_PREDZ;
    float* acc   = ws + WS_ACC;
    _Float16* BE = (_Float16*)(ws + WS_BE);
    _Float16* BD = (_Float16*)(ws + WS_BD);
    float* out   = (float*)d_out;

    zero_acc_kernel<<<1, 32, 0, stream>>>(acc);
    pack_We1_kernel<<<(KCHUNKS * H_VAE * 2 + 255) / 256, 256, 0, stream>>>(We1, BE);
    pack_Wd2_kernel<<<(G_GENES * 4 * 2 + 255) / 256, 256, 0, stream>>>(Wd2, BD);
    encoder_kernel<<<N_CELLS / 16, 256, 0, stream>>>(s, BE, be1, Wet, bet, Wez, bez, zsb, Ts, 1);
    encoder_kernel<<<N_CELLS / 16, 256, 0, stream>>>(u, BE, be1, Wet, bet, Wez, bez, zub, Ts, 0);
    sort_kernel<<<1, 1024, 0, stream>>>(Ts, Tss, idxs);
    ode_kernel<<<1, 32, 0, stream>>>(Tss, idxs, zsb, Wo1, bo1, Wo2, bo2, predz);
    decoder_kernel<<<N_CELLS / 16, 256, 0, stream>>>(zsb, s, Wd1, bd1, BD, bd2, acc, 0);
    decoder_kernel<<<N_CELLS / 16, 256, 0, stream>>>(zub, u, Wd1, bd1, BD, bd2, acc, 1);
    zdiv_kernel<<<N_CELLS / 8, 256, 0, stream>>>(predz, idxs, zsb, zub, Wo1, bo1, Wo2, bo2, beta, lam, acc);
    finalize_kernel<<<1, 32, 0, stream>>>(acc, out);
}
